// ProjectedAdaptiveLogSoftmax_41094247088744
// MI455X (gfx1250) — matrix operand order repaired and checked by reading, not compiled, not run on hardware
//
#include <hip/hip_runtime.h>
#include <hip/hip_bf16.h>
#include <cstdint>
#include <cstddef>

// ---------------------------------------------------------------------------
// Types for CDNA5 WMMA (wave32): v_wmma_f32_16x16x32_bf16
// ---------------------------------------------------------------------------
typedef __attribute__((ext_vector_type(16))) __bf16 v16bf;
typedef __attribute__((ext_vector_type(8)))  float  v8f;
typedef __attribute__((ext_vector_type(4)))  int    v4i_;

#define AS1_ __attribute__((address_space(1)))
#define AS3_ __attribute__((address_space(3)))

union AFrag { v16bf v; uint4 q[2]; };
union BFrag { v16bf v; __bf16 e[16]; };

// Async global->LDS copy path (CDNA5): guarded so the file still compiles on
// toolchains without the builtin.
#if defined(__gfx1250__) && __has_builtin(__builtin_amdgcn_global_load_async_to_lds_b128) && __has_builtin(__builtin_amdgcn_s_wait_asynccnt)
#define USE_ASYNC_LDS 1
#endif

// B fragment: lane holds column n, K-range [kstart, kstart+16). Row-major f32
// weight row; convert to bf16 with native casts (v_cvt_pk_bf16_f32).
// Whole 16-chunk is either fully valid or fully OOB (Kreal, kstart are
// multiples of 16); only the K=16 tail ever takes the zero path.
__device__ __forceinline__ void loadBfrag(BFrag& f, const float* __restrict__ wrow,
                                          int kstart, int Kreal) {
    if (kstart < Kreal) {
        const float4* p = reinterpret_cast<const float4*>(wrow + kstart);
#pragma unroll
        for (int q = 0; q < 4; ++q) {
            float4 x = p[q];
            f.e[4 * q + 0] = (__bf16)x.x;
            f.e[4 * q + 1] = (__bf16)x.y;
            f.e[4 * q + 2] = (__bf16)x.z;
            f.e[4 * q + 3] = (__bf16)x.w;
        }
    } else {
#pragma unroll
        for (int j = 0; j < 8; ++j)
            reinterpret_cast<uint32_t*>(&f)[j] = 0u;
    }
}

// ---------------------------------------------------------------------------
// prep: hidden f32 -> bf16 copy; zero proj3 pad buffer and f32 accumulators
// ---------------------------------------------------------------------------
__global__ void prep_kernel(const float* __restrict__ h,
                            __bf16* __restrict__ hbf,
                            __bf16* __restrict__ p3,
                            float* __restrict__ fzero) {
    int idx = blockIdx.x * blockDim.x + threadIdx.x;   // 0 .. 1024*1024-1
    hbf[idx] = (__bf16)h[idx];
    if (idx < 1024 * 32) p3[idx] = (__bf16)0.0f;       // proj3 zero pad (K=32)
    if (idx < 7 * 1024)  fzero[idx] = 0.0f;            // se[4][1024]+3 gather bufs
}

// ---------------------------------------------------------------------------
// proj: proj_i = hidden @ P_i.T  (M=1024, N=d_i, K=1024), store bf16.
// One wave per block; grid = (d_i/16, 64 token tiles); d_i is a multiple of 16
// so every column is valid.
// ---------------------------------------------------------------------------
__global__ __launch_bounds__(32) void proj_kernel(
    const __bf16* __restrict__ A, const float* __restrict__ P,
    __bf16* __restrict__ outp, int ldo) {
    const int lane = threadIdx.x & 31;
    const int half = lane >> 4;
    const int ln   = lane & 15;
    const int ttile = blockIdx.y;
    const int nb    = blockIdx.x * 16;

    const float*  wrow = P + (size_t)(nb + ln) * 1024;
    const __bf16* arow = A + (size_t)(ttile * 16 + ln) * 1024;

    v8f acc = {};
    for (int k0 = 0; k0 < 1024; k0 += 32) {
        AFrag a;
        const __bf16* ap = arow + k0 + 16 * half;
        a.q[0] = *reinterpret_cast<const uint4*>(ap);
        a.q[1] = *reinterpret_cast<const uint4*>(ap + 8);
        BFrag b; loadBfrag(b, wrow, k0 + 16 * half, 1024);
        acc = __builtin_amdgcn_wmma_f32_16x16x32_bf16(
            false, a.v, false, b.v, (short)0, acc, false, false);
    }
#pragma unroll
    for (int r = 0; r < 8; ++r)
        outp[(size_t)(ttile * 16 + 8 * half + r) * ldo + nb + ln] = (__bf16)acc[r];
}

// ---------------------------------------------------------------------------
// logits: fused GEMM + bias + streaming sum-exp + target/cluster gathers.
// Block = 128 threads (4 waves). Each wave: 16 tokens x 64 cols (4 WMMA
// accumulators); block covers 256 columns. grid = (ceil(N/256), 64).
// A tile (16 x lda bf16, lda == padded K) staged in dynamic LDS once per
// block via async global->LDS copy (fallback: plain LDS stores).
//   seg==0 (head): W1 rows [0,split), W2 (clusters) rows [split,Ntot).
//   seg>=1 (tail): split==Ntot, gathers target-start when cid==seg.
// ---------------------------------------------------------------------------
__global__ __launch_bounds__(128) void logits_kernel(
    const __bf16* __restrict__ A, int lda, int Kreal,
    const float* __restrict__ W1, const float* __restrict__ bv1, int split,
    const float* __restrict__ W2, const float* __restrict__ bv2, int Ntot, int ldb,
    const int* __restrict__ target, int seg, int segStart,
    float* __restrict__ sumexp, float* __restrict__ gtgt, float* __restrict__ gclus) {
    const int wave = threadIdx.x >> 5;
    const int lane = threadIdx.x & 31;
    const int half = lane >> 4;
    const int ln   = lane & 15;
    const int ttile   = blockIdx.y;
    const int colBase = blockIdx.x * 256 + wave * 64;

    extern __shared__ __align__(16) unsigned char smem_raw[];
    __bf16* sA = reinterpret_cast<__bf16*>(smem_raw);   // [16][lda]

    __shared__ int   s_tgt[16];
    __shared__ int   s_cid[16];
    __shared__ float s_rows[4][16];

    if (threadIdx.x < 16) {
        int t = target[ttile * 16 + threadIdx.x];
        s_tgt[threadIdx.x] = t;
        s_cid[threadIdx.x] = (t >= 20000) + (t >= 40000) + (t >= 200000);
    }

    // Stage the contiguous 16-row A tile (rows have stride lda == padded K)
    {
        const int nvec = 16 * lda / 8;                  // uint4 = 8 bf16
        const uint4* gsrc = reinterpret_cast<const uint4*>(A + (size_t)ttile * 16 * lda);
        uint4* ldst = reinterpret_cast<uint4*>(sA);
#if defined(USE_ASYNC_LDS)
        for (int i = threadIdx.x; i < nvec; i += 128)
            __builtin_amdgcn_global_load_async_to_lds_b128(
                (AS1_ v4i_*)(gsrc + i), (AS3_ v4i_*)(ldst + i), 0, 0);
        __builtin_amdgcn_s_wait_asynccnt(0);
#else
        for (int i = threadIdx.x; i < nvec; i += 128)
            ldst[i] = gsrc[i];
#endif
    }
    __syncthreads();

    // Per-lane column pointers / bias; OOB columns clamp to the last valid row
    // (their contribution is masked in the epilogue via valid[]).
    const float* wrow[4];
    float bias[4];
    bool  valid[4];
#pragma unroll
    for (int ct = 0; ct < 4; ++ct) {
        int n = colBase + ct * 16 + ln;
        valid[ct] = (n < Ntot);
        int nc = valid[ct] ? n : (Ntot - 1);
        if (nc < split) { wrow[ct] = W1 + (size_t)nc * ldb; bias[ct] = bv1[nc]; }
        else { wrow[ct] = W2 + (size_t)(nc - split) * ldb; bias[ct] = bv2[nc - split]; }
    }

    const int Kpad = (Kreal + 31) & ~31;
    v8f acc[4];
    {
        v8f z = {};
#pragma unroll
        for (int ct = 0; ct < 4; ++ct) acc[ct] = z;
    }

    for (int k0 = 0; k0 < Kpad; k0 += 32) {
        AFrag a;
        const __bf16* ap = sA + (size_t)ln * lda + k0 + 16 * half;  // ds_load_b128
        a.q[0] = *reinterpret_cast<const uint4*>(ap);
        a.q[1] = *reinterpret_cast<const uint4*>(ap + 8);
        if (k0 + 160 < Kreal) {
#pragma unroll
            for (int ct = 0; ct < 4; ++ct)
                __builtin_prefetch(wrow[ct] + k0 + 128, 0, 1);     // global_prefetch_b8
        }
#pragma unroll
        for (int ct = 0; ct < 4; ++ct) {
            BFrag b; loadBfrag(b, wrow[ct], k0 + 16 * half, Kreal);
            acc[ct] = __builtin_amdgcn_wmma_f32_16x16x32_bf16(
                false, a.v, false, b.v, (short)0, acc[ct], false, false);
        }
    }

    // Epilogue: bias add, exp accumulation and gathers.
    // C layout: lane holds column n = ln, rows m = 8*half + r (r = VGPR idx).
    float rowpart[8];
#pragma unroll
    for (int r = 0; r < 8; ++r) rowpart[r] = 0.0f;

#pragma unroll
    for (int ct = 0; ct < 4; ++ct) {
        if (!valid[ct]) continue;
        const int n = colBase + ct * 16 + ln;
#pragma unroll
        for (int r = 0; r < 8; ++r) {
            float lg = acc[ct][r] + bias[ct];
            rowpart[r] += __expf(lg);
            const int m   = 8 * half + r;
            const int row = ttile * 16 + m;
            if (seg == 0) {
                if (s_cid[m] == 0 && n == s_tgt[m])       gtgt[row]  = lg;
                if (n >= split && s_cid[m] == (Ntot - n)) gclus[row] = lg;
            } else {
                if (s_cid[m] == seg && (s_tgt[m] - segStart) == n) gtgt[row] = lg;
            }
        }
    }

    // Reduce Σexp across the 16 lanes of each half (half h owns rows 8h..8h+7)
#pragma unroll
    for (int r = 0; r < 8; ++r) {
#pragma unroll
        for (int off = 1; off < 16; off <<= 1)
            rowpart[r] += __shfl_xor(rowpart[r], off, 32);
    }
    if (ln == 0) {
#pragma unroll
        for (int r = 0; r < 8; ++r) s_rows[wave][8 * half + r] = rowpart[r];
    }
    __syncthreads();
    if (threadIdx.x < 16) {
        float s = s_rows[0][threadIdx.x] + s_rows[1][threadIdx.x] +
                  s_rows[2][threadIdx.x] + s_rows[3][threadIdx.x];
        atomicAdd(&sumexp[ttile * 16 + threadIdx.x], s);
    }
}

// ---------------------------------------------------------------------------
// finalize: per-token NLL assembly
// ---------------------------------------------------------------------------
__global__ void finalize_kernel(const int* __restrict__ target,
                                const float* __restrict__ se,
                                const float* __restrict__ tgtP,
                                const float* __restrict__ clusL,
                                const float* __restrict__ tgtT,
                                float* __restrict__ out) {
    int t = blockIdx.x * blockDim.x + threadIdx.x;
    if (t >= 1024) return;
    int tg  = target[t];
    int cid = (tg >= 20000) + (tg >= 40000) + (tg >= 200000);
    float lseH = __logf(se[t]);
    float lp;
    if (cid == 0) {
        lp = tgtP[t] - lseH;
    } else {
        lp = (clusL[t] - lseH) + (tgtT[t] - __logf(se[cid * 1024 + t]));
    }
    out[t] = -lp;
}

// ---------------------------------------------------------------------------
// launch
// ---------------------------------------------------------------------------
extern "C" void kernel_launch(void* const* d_in, const int* in_sizes, int n_in,
                              void* d_out, int out_size, void* d_ws, size_t ws_size,
                              hipStream_t stream) {
    const float* hidden  = (const float*)d_in[0];
    const int*   target  = (const int*)  d_in[1];
    const float* W_head  = (const float*)d_in[2];
    const float* b_head  = (const float*)d_in[3];
    const float* W_cl    = (const float*)d_in[4];
    const float* b_cl    = (const float*)d_in[5];
    const float* P1      = (const float*)d_in[6];
    const float* W1      = (const float*)d_in[7];
    const float* b1      = (const float*)d_in[8];
    const float* P2      = (const float*)d_in[9];
    const float* W2      = (const float*)d_in[10];
    const float* b2      = (const float*)d_in[11];
    const float* P3      = (const float*)d_in[12];
    const float* W3      = (const float*)d_in[13];
    const float* b3      = (const float*)d_in[14];

    uint8_t* ws = (uint8_t*)d_ws;
    __bf16* hbf = (__bf16*)ws;                                   // 1024x1024
    __bf16* p1  = (__bf16*)(ws + 2097152);                       // 1024x256
    __bf16* p2  = (__bf16*)(ws + 2097152 + 524288);              // 1024x64
    __bf16* p3  = (__bf16*)(ws + 2097152 + 524288 + 131072);     // 1024x32 (padded)
    float* se    = (float*)(ws + 2097152 + 524288 + 131072 + 65536); // [4][1024]
    float* tgtP  = se + 4096;   // head target logit (cluster 0)
    float* clusL = tgtP + 1024; // head cluster-column logit
    float* tgtT  = clusL + 1024;// tail target logit

    prep_kernel<<<4096, 256, 0, stream>>>(hidden, hbf, p3, se);

    proj_kernel<<<dim3(16, 64), 32, 0, stream>>>(hbf, P1, p1, 256);
    proj_kernel<<<dim3(4, 64),  32, 0, stream>>>(hbf, P2, p2,  64);
    proj_kernel<<<dim3(1, 64),  32, 0, stream>>>(hbf, P3, p3,  32);

    // head: N = 20003 (20000 vocab + 3 cluster cols), K = 1024
    logits_kernel<<<dim3(79, 64), 128, 16 * 1024 * 2, stream>>>(
        hbf, 1024, 1024, W_head, b_head, 20000, W_cl, b_cl, 20003, 1024,
        target, 0, 0, se, tgtP, clusL);
    // tail1: N = 20000, K = 256
    logits_kernel<<<dim3(79, 64), 128, 16 * 256 * 2, stream>>>(
        p1, 256, 256, W1, b1, 20000, W1, b1, 20000, 256,
        target, 1, 20000, se + 1024, tgtT, clusL);
    // tail2: N = 160000, K = 64
    logits_kernel<<<dim3(625, 64), 128, 16 * 64 * 2, stream>>>(
        p2, 64, 64, W2, b2, 160000, W2, b2, 160000, 64,
        target, 2, 40000, se + 2048, tgtT, clusL);
    // tail3: N = 67735, K = 16 (A padded to 32)
    logits_kernel<<<dim3(265, 64), 128, 16 * 32 * 2, stream>>>(
        p3, 32, 16, W3, b3, 67735, W3, b3, 67735, 16,
        target, 3, 200000, se + 3072, tgtT, clusL);

    finalize_kernel<<<4, 256, 0, stream>>>(target, se, tgtP, clusL, tgtT,
                                           (float*)d_out);
}
